// SelfAttention_45921790329648
// MI455X (gfx1250) — matrix operand order repaired and checked by reading, not compiled, not run on hardware
//
#include <hip/hip_runtime.h>

// ---------------------------------------------------------------------------
// Types for CDNA5 WMMA (gfx1250, wave32)
// ---------------------------------------------------------------------------
typedef __bf16 bf16_t;
typedef bf16_t v16bf __attribute__((ext_vector_type(16)));
typedef bf16_t v8bf  __attribute__((ext_vector_type(8)));
typedef float  v8f   __attribute__((ext_vector_type(8)));

#define NEG_INF (-__builtin_inff())

// Model constants
#define BATCH   4
#define SEQ     2048
#define CMODEL  1024
#define NHEADS  16
#define DHEAD   64
#define NTOK    (BATCH * SEQ)          // 8192
#define C3      (3 * CMODEL)           // 3072

// ---------------------------------------------------------------------------
// Fragment loaders.
//
// A-matrix 16x32 bf16 (ISA 7.12.2 table): lane l holds row m = l&15;
// khalf = l>>4 selects the 8-element sub-runs:
//   elements 0..7  -> K = khalf*8 + 0..7
//   elements 8..15 -> K = 16 + khalf*8 + 0..7
// => two contiguous 16B loads from a K-contiguous row.
//
// B-matrix 32x16 bf16 (same K order as A): lane l holds col n = l&15;
//   elements 0..7  -> K = khalf*8 + 0..7
//   elements 8..15 -> K = 16 + khalf*8 + 0..7
// ---------------------------------------------------------------------------
__device__ __forceinline__ v16bf load_a_frag(const bf16_t* rowbase, int khalf) {
    const v8bf lo = *(const v8bf*)(rowbase + khalf * 8);
    const v8bf hi = *(const v8bf*)(rowbase + 16 + khalf * 8);
    v16bf f;
#pragma unroll
    for (int i = 0; i < 8; ++i) { f[i] = lo[i]; f[i + 8] = hi[i]; }
    return f;
}

__device__ __forceinline__ v16bf load_b_frag(const bf16_t* rowbase, int khalf) {
    const bf16_t* p = rowbase + khalf * 8;
    const v8bf lo = *(const v8bf*)(p);
    const v8bf hi = *(const v8bf*)(p + 16);
    v16bf f;
#pragma unroll
    for (int i = 0; i < 8; ++i) { f[i] = lo[i]; f[i + 8] = hi[i]; }
    return f;
}

__device__ __forceinline__ v8f wmma_bf16(v16bf a, v16bf b, v8f c) {
    return __builtin_amdgcn_wmma_f32_16x16x32_bf16(
        /*neg_a=*/false, a, /*neg_b=*/false, b,
        /*c_mod=*/(short)0, c, /*reuse_a=*/false, /*reuse_b=*/false);
}

// Row reductions across the 16 lanes of a half-wave (C-fragment rows live
// across lanes 0..15 / 16..31 of one VGPR).
__device__ __forceinline__ float rowmax16(float v) {
    v = fmaxf(v, __shfl_xor(v, 1));
    v = fmaxf(v, __shfl_xor(v, 2));
    v = fmaxf(v, __shfl_xor(v, 4));
    v = fmaxf(v, __shfl_xor(v, 8));
    return v;
}
__device__ __forceinline__ float rowsum16(float v) {
    v += __shfl_xor(v, 1);
    v += __shfl_xor(v, 2);
    v += __shfl_xor(v, 4);
    v += __shfl_xor(v, 8);
    return v;
}

// ---------------------------------------------------------------------------
// Stage 0: fp32 -> bf16 conversions / transposes
// ---------------------------------------------------------------------------
__global__ void cvt_x_kernel(const float* __restrict__ x, bf16_t* __restrict__ xb, int n) {
    for (int i = blockIdx.x * blockDim.x + threadIdx.x; i < n; i += gridDim.x * blockDim.x)
        xb[i] = (bf16_t)x[i];
}

// wT[n*K + k] = w[k*N + n]
__global__ void cvt_transpose_kernel(const float* __restrict__ w, bf16_t* __restrict__ wT,
                                     int N, int K) {
    int total = N * K;
    for (int i = blockIdx.x * blockDim.x + threadIdx.x; i < total; i += gridDim.x * blockDim.x) {
        int n = i / K;
        int k = i - n * K;
        wT[i] = (bf16_t)w[(size_t)k * N + n];
    }
}

// ---------------------------------------------------------------------------
// Stage 1: QKV GEMM.  qkv = xb[8192,1024] @ wqkvT^T + b_qkv
// One wave computes a 32x64 output tile (2 A-frags x 4 B-frags -> 8 WMMAs
// per 32-wide K step; B fragments reused across both M sub-tiles to double
// arithmetic intensity vs a 16-row tile).  Results scattered to
//   q  [B,H,T,Dh] ; k [B,H,T,Dh] ; vT [B,H,Dh,T]
// grid = (48 n-tiles, 256 m-tiles), block = 32
// ---------------------------------------------------------------------------
__global__ void qkv_gemm_kernel(const bf16_t* __restrict__ xb,
                                const bf16_t* __restrict__ wT,
                                const float*  __restrict__ bias,
                                bf16_t* __restrict__ qbuf,
                                bf16_t* __restrict__ kbuf,
                                bf16_t* __restrict__ vTbuf) {
    const int lane  = threadIdx.x;
    const int nloc  = lane & 15;
    const int khalf = lane >> 4;
    const int n0 = blockIdx.x * 64;
    const int m0 = blockIdx.y * 32;

    v8f acc[2][4] = {};
    const bf16_t* arow0 = xb + (size_t)(m0 + nloc) * CMODEL;
    const bf16_t* arow1 = xb + (size_t)(m0 + 16 + nloc) * CMODEL;

    for (int k0 = 0; k0 < CMODEL; k0 += 32) {
        // prefetch next K tile (global_prefetch_b8; speculative, cheap)
        __builtin_prefetch(arow0 + k0 + 128, 0, 1);
        __builtin_prefetch(arow1 + k0 + 128, 0, 1);
        v16bf a0 = load_a_frag(arow0 + k0, khalf);
        v16bf a1 = load_a_frag(arow1 + k0, khalf);
#pragma unroll
        for (int f = 0; f < 4; ++f) {
            const bf16_t* brow = wT + (size_t)(n0 + f * 16 + nloc) * CMODEL + k0;
            __builtin_prefetch(brow + 128, 0, 1);
            v16bf b = load_b_frag(brow, khalf);
            acc[0][f] = wmma_bf16(a0, b, acc[0][f]);
            acc[1][f] = wmma_bf16(a1, b, acc[1][f]);
        }
    }

    const int sec = n0 / CMODEL;           // 0:q 1:k 2:v (uniform per block)
    const int h   = (n0 % CMODEL) / DHEAD; // uniform per block
#pragma unroll
    for (int mi = 0; mi < 2; ++mi) {
#pragma unroll
        for (int f = 0; f < 4; ++f) {
            const int d  = f * 16 + nloc;
            const float bv = bias[n0 + f * 16 + nloc];
#pragma unroll
            for (int r = 0; r < 8; ++r) {
                const int tok = m0 + mi * 16 + r + 8 * khalf;
                const int b   = tok >> 11;       // / SEQ
                const int t   = tok & (SEQ - 1);
                const float val = acc[mi][f][r] + bv;
                const size_t bh = (size_t)(b * NHEADS + h);
                if (sec == 0)
                    qbuf[(bh * SEQ + t) * DHEAD + d] = (bf16_t)val;
                else if (sec == 1)
                    kbuf[(bh * SEQ + t) * DHEAD + d] = (bf16_t)val;
                else
                    vTbuf[(bh * DHEAD + d) * SEQ + t] = (bf16_t)val;
            }
        }
    }
}

// ---------------------------------------------------------------------------
// Stage 2: flash attention.  One wave per (b*h, 16-query tile).
// grid = (64, 128), block = 32.  Key tiles of 32, online softmax.
// ---------------------------------------------------------------------------
__global__ void attn_kernel(const bf16_t* __restrict__ qbuf,
                            const bf16_t* __restrict__ kbuf,
                            const bf16_t* __restrict__ vTbuf,
                            bf16_t* __restrict__ y) {
    __shared__ bf16_t Pl[16 * 32];   // P tile staging: C-layout -> A-layout

    const int lane  = threadIdx.x;
    const int nloc  = lane & 15;
    const int khalf = lane >> 4;
    const int bh = blockIdx.x;            // 0..63
    const int b  = bh / NHEADS;
    const int h  = bh % NHEADS;
    const int qt = blockIdx.y;            // 0..127
    const int q0 = qt * 16;

    // Preload the two Q A-fragments (d = 0..31, 32..63) for this query tile.
    const bf16_t* qrow = qbuf + ((size_t)bh * SEQ + q0 + nloc) * DHEAD;
    const v16bf qa0 = load_a_frag(qrow, khalf);
    const v16bf qa1 = load_a_frag(qrow + 32, khalf);

    v8f o[4] = {};
    float mrow[8], lrow[8];
#pragma unroll
    for (int r = 0; r < 8; ++r) { mrow[r] = NEG_INF; lrow[r] = 0.0f; }

    const float scale = 0.125f;           // 1/sqrt(64)
    const int nkeys = q0 + 16;            // causal
    const int nkt = (nkeys + 31) / 32;

    for (int kt = 0; kt < nkt; ++kt) {
        const int key0 = kt * 32;

        const bf16_t* krow0 = kbuf + ((size_t)bh * SEQ + key0 + nloc) * DHEAD;
        const bf16_t* krow1 = kbuf + ((size_t)bh * SEQ + key0 + 16 + nloc) * DHEAD;
        // prefetch next key tile's K rows
        __builtin_prefetch(krow0 + 32 * DHEAD, 0, 1);
        __builtin_prefetch(krow1 + 32 * DHEAD, 0, 1);

        // ---- S = Q K^T for 16 queries x 32 keys (two 16-wide C fragments)
        v8f c0 = {}, c1 = {};
        c0 = wmma_bf16(qa0, load_b_frag(krow0, khalf), c0);
        c0 = wmma_bf16(qa1, load_b_frag(krow0 + 32, khalf), c0);
        c1 = wmma_bf16(qa0, load_b_frag(krow1, khalf), c1);
        c1 = wmma_bf16(qa1, load_b_frag(krow1 + 32, khalf), c1);

        // ---- scale + causal mask + online softmax update
        float p0[8], p1[8], alpha[8];
#pragma unroll
        for (int r = 0; r < 8; ++r) {
            const int qg = q0 + r + 8 * khalf;
            float s0 = c0[r] * scale;
            float s1 = c1[r] * scale;
            if (key0 + nloc > qg)      s0 = NEG_INF;
            if (key0 + 16 + nloc > qg) s1 = NEG_INF;
            const float tmax = rowmax16(fmaxf(s0, s1));
            const float nm = fmaxf(mrow[r], tmax);
            alpha[r] = __expf(mrow[r] - nm);    // 0 on first tile (mrow=-inf)
            p0[r] = __expf(s0 - nm);            // masked -> exp(-inf)=0
            p1[r] = __expf(s1 - nm);
            lrow[r] = lrow[r] * alpha[r] + rowsum16(p0[r] + p1[r]);
            mrow[r] = nm;
        }
#pragma unroll
        for (int f = 0; f < 4; ++f)
#pragma unroll
            for (int r = 0; r < 8; ++r) o[f][r] *= alpha[r];

        // ---- stage P (C-layout) into LDS, reread in A-layout
#pragma unroll
        for (int r = 0; r < 8; ++r) {
            const int m = r + 8 * khalf;
            Pl[m * 32 + nloc]      = (bf16_t)p0[r];
            Pl[m * 32 + 16 + nloc] = (bf16_t)p1[r];
        }
        __syncthreads();
        const v16bf pa = load_a_frag(&Pl[nloc * 32], khalf);

        // ---- O += P V   (4 d-fragments of 16)
#pragma unroll
        for (int f = 0; f < 4; ++f) {
            const bf16_t* vrow = vTbuf + ((size_t)bh * DHEAD + f * 16 + nloc) * SEQ + key0;
            __builtin_prefetch(vrow + 32, 0, 1);
            o[f] = wmma_bf16(pa, load_b_frag(vrow, khalf), o[f]);
        }
        __syncthreads();
    }

    // ---- normalize and write y[B,T,C] (C = h*64 + d), bf16 for final GEMM
    float inv[8];
#pragma unroll
    for (int r = 0; r < 8; ++r) inv[r] = 1.0f / lrow[r];
#pragma unroll
    for (int f = 0; f < 4; ++f)
#pragma unroll
        for (int r = 0; r < 8; ++r) {
            const int t = q0 + r + 8 * khalf;
            const int c = h * DHEAD + f * 16 + nloc;
            y[((size_t)b * SEQ + t) * CMODEL + c] = (bf16_t)(o[f][r] * inv[r]);
        }
}

// ---------------------------------------------------------------------------
// Stage 3: output GEMM.  out = y[8192,1024] @ woutT^T + b_out  (fp32 out)
// 32x64 tile per wave, same structure as stage 1.
// grid = (16 n-tiles, 256 m-tiles), block = 32
// ---------------------------------------------------------------------------
__global__ void out_gemm_kernel(const bf16_t* __restrict__ y,
                                const bf16_t* __restrict__ wT,
                                const float*  __restrict__ bias,
                                float* __restrict__ out) {
    const int lane  = threadIdx.x;
    const int nloc  = lane & 15;
    const int khalf = lane >> 4;
    const int n0 = blockIdx.x * 64;
    const int m0 = blockIdx.y * 32;

    v8f acc[2][4] = {};
    const bf16_t* arow0 = y + (size_t)(m0 + nloc) * CMODEL;
    const bf16_t* arow1 = y + (size_t)(m0 + 16 + nloc) * CMODEL;

    for (int k0 = 0; k0 < CMODEL; k0 += 32) {
        __builtin_prefetch(arow0 + k0 + 128, 0, 1);
        __builtin_prefetch(arow1 + k0 + 128, 0, 1);
        v16bf a0 = load_a_frag(arow0 + k0, khalf);
        v16bf a1 = load_a_frag(arow1 + k0, khalf);
#pragma unroll
        for (int f = 0; f < 4; ++f) {
            const bf16_t* brow = wT + (size_t)(n0 + f * 16 + nloc) * CMODEL + k0;
            __builtin_prefetch(brow + 128, 0, 1);
            v16bf b = load_b_frag(brow, khalf);
            acc[0][f] = wmma_bf16(a0, b, acc[0][f]);
            acc[1][f] = wmma_bf16(a1, b, acc[1][f]);
        }
    }

#pragma unroll
    for (int mi = 0; mi < 2; ++mi) {
#pragma unroll
        for (int f = 0; f < 4; ++f) {
            const float bv = bias[n0 + f * 16 + nloc];
#pragma unroll
            for (int r = 0; r < 8; ++r) {
                const int tok = m0 + mi * 16 + r + 8 * khalf;
                out[(size_t)tok * CMODEL + n0 + f * 16 + nloc] = acc[mi][f][r] + bv;
            }
        }
    }
}

// ---------------------------------------------------------------------------
// Host-side launcher
// ---------------------------------------------------------------------------
extern "C" void kernel_launch(void* const* d_in, const int* in_sizes, int n_in,
                              void* d_out, int out_size, void* d_ws, size_t ws_size,
                              hipStream_t stream) {
    (void)in_sizes; (void)n_in; (void)out_size; (void)ws_size;

    const float* x     = (const float*)d_in[0];  // [4,2048,1024]
    const float* w_qkv = (const float*)d_in[1];  // [1024,3072]
    const float* b_qkv = (const float*)d_in[2];  // [3072]
    const float* w_out = (const float*)d_in[3];  // [1024,1024]
    const float* b_out = (const float*)d_in[4];  // [1024]
    float* out = (float*)d_out;                  // [4,2048,1024]

    // Workspace carve-out (all 256B aligned)
    size_t off = 0;
    auto carve = [&](size_t bytes) -> char* {
        char* p = (char*)d_ws + off;
        off += (bytes + 255) & ~(size_t)255;
        return p;
    };
    bf16_t* xb     = (bf16_t*)carve((size_t)NTOK * CMODEL * sizeof(bf16_t));
    bf16_t* wqkvT  = (bf16_t*)carve((size_t)C3 * CMODEL * sizeof(bf16_t));
    bf16_t* woutT  = (bf16_t*)carve((size_t)CMODEL * CMODEL * sizeof(bf16_t));
    bf16_t* qbuf   = (bf16_t*)carve((size_t)BATCH * NHEADS * SEQ * DHEAD * sizeof(bf16_t));
    bf16_t* kbuf   = (bf16_t*)carve((size_t)BATCH * NHEADS * SEQ * DHEAD * sizeof(bf16_t));
    bf16_t* vTbuf  = (bf16_t*)carve((size_t)BATCH * NHEADS * DHEAD * SEQ * sizeof(bf16_t));
    bf16_t* ybuf   = (bf16_t*)carve((size_t)NTOK * CMODEL * sizeof(bf16_t));

    // Stage 0: conversions
    cvt_x_kernel<<<4096, 256, 0, stream>>>(x, xb, NTOK * CMODEL);
    cvt_transpose_kernel<<<4096, 256, 0, stream>>>(w_qkv, wqkvT, C3, CMODEL);
    cvt_transpose_kernel<<<2048, 256, 0, stream>>>(w_out, woutT, CMODEL, CMODEL);

    // Stage 1: QKV projection (WMMA, 32x64 tiles)
    qkv_gemm_kernel<<<dim3(C3 / 64, NTOK / 32), 32, 0, stream>>>(
        xb, wqkvT, b_qkv, qbuf, kbuf, vTbuf);

    // Stage 2: causal flash attention (WMMA)
    attn_kernel<<<dim3(BATCH * NHEADS, SEQ / 16), 32, 0, stream>>>(
        qbuf, kbuf, vTbuf, ybuf);

    // Stage 3: output projection (WMMA, 32x64 tiles)
    out_gemm_kernel<<<dim3(CMODEL / 64, NTOK / 32), 32, 0, stream>>>(
        ybuf, woutT, b_out, out);
}
